// RefineBlock_PWCLO_62062277427562
// MI455X (gfx1250) — compile-verified, matrix operand order checked by reading, not run on hardware
//
#include <hip/hip_runtime.h>
#include <hip/hip_bf16.h>
#include <math.h>

// ---------------------------------------------------------------------------
// RefineBlock (PWCLO) for MI455X / gfx1250, wave32 + WMMA f16 (f32 accum),
// with async global->LDS weight staging (ASYNCcnt path).
// B=2, N=4096, C=64, K=16 neighbors in both stages.
// ---------------------------------------------------------------------------

typedef __attribute__((ext_vector_type(16))) _Float16 v16h;
typedef __attribute__((ext_vector_type(8)))  float    v8f;
typedef __attribute__((ext_vector_type(4)))  float    v4f;

#define B_   2
#define N_   4096
#define C_   64

// packed-weight offsets (in halves) for the 9 layers:
// 0 mlp1_0(138->128,K160) 1 mlp1_1(128->64) 2 mlp1_2(64->64) 3 enc1(10->64,K32)
// 4 mlp2_0(128->128) 5 mlp2_1(128->64) 6 enc2(10->64,K32)
// 7 mlp3_0(192->128) 8 mlp3_1(128->64)
constexpr unsigned kWOFF[9] = {0u, 20480u, 28672u, 32768u, 34816u, 51200u, 59392u, 61440u, 86016u};

// stage-local LDS weight blocks (layers 0..5 and 6..8 are contiguous)
constexpr unsigned kS1_W_BYTES  = 59392u * 2u;   // 118784 B, = 29 * 4096
constexpr unsigned kS2_W_BYTES  = 34816u * 2u;   // 69632 B,  = 17 * 4096
constexpr unsigned kS1_W_LDSOFF = 40960u + 32768u + 32768u + 32768u;  // 139264
constexpr unsigned kS2_W_LDSOFF = 8192u + 49152u + 32768u + 32768u;   // 122880
constexpr unsigned kSMEM1 = kS1_W_LDSOFF + kS1_W_BYTES;               // 258048
constexpr unsigned kSMEM2 = kS2_W_LDSOFF + kS2_W_BYTES;               // 192512

// workspace byte offsets
constexpr size_t WS_WPACK = 0;
constexpr size_t WS_ST    = 188416;
constexpr size_t WS_IDXQ  = 225280;                       // B*N*16 ints
constexpr size_t WS_IDXS  = 749568;                       // B*N*16 ints
constexpr size_t WS_FEAT1 = 1273856;                      // B*N*64 floats

// ---------------------------------------------------------------------------
// Async global->LDS staging (CDNA5 ASYNCcnt engine). Each thread moves 16B
// per round; `bytes` must be a multiple of 4096 (256 threads * 16B).
// GVS addressing: mem = SADDR(64b SGPR) + VADDR(32b VGPR) ; LDS dest offset
// is carried in the VDST VGPR.
// ---------------------------------------------------------------------------
__device__ __forceinline__ void async_stage_weights(unsigned ldsDstBase,
                                                    const _Float16* gsrc,
                                                    int bytes, int tid) {
  unsigned long long sbase = (unsigned long long)(const void*)gsrc;
  for (int off = tid * 16; off < bytes; off += 256 * 16) {
    unsigned ldsa = ldsDstBase + (unsigned)off;
    unsigned voff = (unsigned)off;
    asm volatile("global_load_async_to_lds_b128 %0, %1, %2"
                 :
                 : "v"(ldsa), "v"(voff), "s"(sbase)
                 : "memory");
  }
}

__device__ __forceinline__ void wait_async_lds() {
  asm volatile("s_wait_asynccnt 0x0" ::: "memory");
}

// ---------------------------------------------------------------------------
// Weight prep: transpose W[Cout][Cin] into per-lane WMMA B fragments (f16,
// K zero-padded) and fold BatchNorm into per-channel scale/bias.
// Packed layout: frag f = kt*Ntiles+nt holds 32 lanes x 16 halves; lane l
// (n = l&15, kh = l>>4) halves e: K = kt*32 + (e<8 ? kh*8+e : 16+kh*8+e-8).
// ---------------------------------------------------------------------------
__global__ void prep_layer(const float* __restrict__ W, const float* __restrict__ bias,
                           const float* __restrict__ g, const float* __restrict__ beta,
                           const float* __restrict__ rm, const float* __restrict__ rv,
                           int Cin, int Cout, int Kpad,
                           _Float16* __restrict__ Wpack,
                           float* __restrict__ s, float* __restrict__ t) {
  int p = blockIdx.x * blockDim.x + threadIdx.x;
  int total = Kpad * Cout;
  if (p < total) {
    int frag = p >> 9, within = p & 511;
    int lane = within >> 4, e = within & 15;
    int Ntiles = Cout >> 4;
    int kt = frag / Ntiles, nt = frag - kt * Ntiles;
    int kh = lane >> 4, nl = lane & 15;
    int col = (nt << 4) + nl;
    int krel = (e < 8) ? ((kh << 3) + e) : (16 + (kh << 3) + (e - 8));
    int K = (kt << 5) + krel;
    float w = (K < Cin) ? W[col * Cin + K] : 0.0f;
    Wpack[p] = (_Float16)w;
  }
  if (p < Cout) {
    float sc = g[p] * rsqrtf(rv[p] + 1e-5f);
    s[p] = sc;
    t[p] = (bias[p] - rm[p]) * sc + beta[p];
  }
}

// ---------------------------------------------------------------------------
// Brute-force KNN (top-16 smallest squared distance). 256 queries per block.
// ---------------------------------------------------------------------------
__global__ void knn_kernel(const float* __restrict__ q, const float* __restrict__ c,
                           int* __restrict__ out) {
  __shared__ float cs[3 * 256];
  const int tid = threadIdx.x;
  const int b = blockIdx.x >> 4;                 // N_/256 = 16 blocks per batch
  const int qi = ((blockIdx.x & 15) << 8) + tid;
  const float* qp = q + ((size_t)b * N_ + qi) * 3;
  const float qx = qp[0], qy = qp[1], qz = qp[2];
  float bd[16]; int bi[16];
#pragma unroll
  for (int i = 0; i < 16; ++i) { bd[i] = 3.0e38f; bi[i] = 0; }
  for (int tile = 0; tile < N_ / 256; ++tile) {
    const float* cp = c + ((size_t)b * N_ + (tile << 8) + tid) * 3;
    cs[tid] = cp[0]; cs[256 + tid] = cp[1]; cs[512 + tid] = cp[2];
    __syncthreads();
    for (int jj = 0; jj < 256; ++jj) {
      float dx = cs[jj] - qx, dy = cs[256 + jj] - qy, dz = cs[512 + jj] - qz;
      float d = dx * dx + dy * dy + dz * dz;
      if (d < bd[15]) {
        float vd = d; int vi = (tile << 8) + jj;
#pragma unroll
        for (int s = 0; s < 16; ++s) {
          bool sw = vd < bd[s];
          float td = sw ? bd[s] : vd; int ti = sw ? bi[s] : vi;
          bd[s] = sw ? vd : bd[s];   bi[s] = sw ? vi : bi[s];
          vd = td; vi = ti;
        }
      }
    }
    __syncthreads();
  }
  int* op = out + ((size_t)b * N_ + qi) * 16;
#pragma unroll
  for (int i = 0; i < 16; ++i) op[i] = bi[i];
}

// ---------------------------------------------------------------------------
// One wave computes a 16-row M-tile of Y[128 x 16*NT] = relu(s*(X@Wp)+t).
// KT/NT are compile-time; A fragments (kt-dependent only) are hoisted into
// registers once, then the fully unrolled WMMA chain streams B fragments
// (one contiguous 32B per-lane ds_load pair) from LDS-staged weights.
// F16OUT selects f16 activation output (with optional column offset) vs f32
// logits output -- resolved at compile time (branch-free epilogue).
// ---------------------------------------------------------------------------
template <int KT, int NT, bool F16OUT>
__device__ __forceinline__ void wave_gemm(
    const _Float16* __restrict__ X, int ldx,
    const _Float16* __restrict__ Wp,            // LDS
    const float* __restrict__ s, const float* __restrict__ t,
    _Float16* __restrict__ Yh, float* __restrict__ Yf, int ldy, int colOff,
    int lane, int wave) {
  const int kh = lane >> 4;
  const int nl = lane & 15;
  const _Float16* arow = X + (wave * 16 + nl) * ldx;
  v16h a[KT];
#pragma unroll
  for (int kt = 0; kt < KT; ++kt) {
    const _Float16* ap = arow + kt * 32 + kh * 8;
    ((v4f*)&a[kt])[0] = *(const v4f*)ap;
    ((v4f*)&a[kt])[1] = *(const v4f*)(ap + 16);
  }
#pragma unroll
  for (int nt = 0; nt < NT; ++nt) {
    v8f acc = {0.f, 0.f, 0.f, 0.f, 0.f, 0.f, 0.f, 0.f};
#pragma unroll
    for (int kt = 0; kt < KT; ++kt) {
      v16h bf = *(const v16h*)(Wp + (((unsigned)(kt * NT + nt) << 5) + lane) * 16);
      acc = __builtin_amdgcn_wmma_f32_16x16x32_f16(false, a[kt], false, bf,
                                                   (short)0, acc, false, false);
    }
    const int oc = (nt << 4) + nl;       // layer output channel
    const float sc = s[oc], bi = t[oc];
    const int col = colOff + oc;
    const int rb = wave * 16 + kh * 8;   // C/D: VGPR r -> M = r + (lane>=16)*8
#pragma unroll
    for (int r = 0; r < 8; ++r) {
      float v = fmaxf(fmaf(acc[r], sc, bi), 0.0f);
      if constexpr (F16OUT) Yh[(rb + r) * ldy + col] = (_Float16)v;
      else                  Yf[(rb + r) * ldy + col] = v;
    }
  }
}

// ---------------------------------------------------------------------------
// Stage 1: cross-cloud grouping + enc1 + mlp1 chain + mlp2 chain + softmax.
// Block = 8 query points x 16 neighbors = 128 rows, 256 threads (8 waves).
// ---------------------------------------------------------------------------
__global__ void __launch_bounds__(256)
stage1_kernel(const float* __restrict__ wxyz, const float* __restrict__ fxyz,
              const float* __restrict__ wpts, const float* __restrict__ fpts,
              const int* __restrict__ idxq,
              const _Float16* __restrict__ Wp, const float* __restrict__ st,
              float* __restrict__ feat1) {
  extern __shared__ char smem[];
  _Float16* X0   = (_Float16*)smem;                            // 128x160 f16 (40960B)
  _Float16* bufA = (_Float16*)(smem + 40960);                  // 128x128 f16 (32768B)
  _Float16* bufC = (_Float16*)(smem + 40960 + 32768);          // 128x128 f16 (enc1|h3)
  float*    bufS = (float*)(smem + 40960 + 32768 + 32768);     // 128x64 f32 logits
  _Float16* bufW = (_Float16*)(smem + kS1_W_LDSOFF);           // staged weights L0..L5
  _Float16* h2   = X0;                                         // 128x64 reuse of X0

  const int tid = threadIdx.x;
  const int lane = tid & 31, wave = tid >> 5;
  const int b = blockIdx.x >> 9;                 // N_/8 = 512 blocks per batch
  const int n0 = (blockIdx.x & 511) << 3;

  // kick off async weight staging; overlaps with the gather-heavy build below
  async_stage_weights(kS1_W_LDSOFF, Wp, (int)kS1_W_BYTES, tid);

  {  // build X0 = [pi_xyz, qi_xyz, diff, euc, pi_pts, qi_pts, 0-pad]
    int row = tid & 127, part = tid >> 7;
    int qq = row >> 4, kq = row & 15;
    int n = n0 + qq;
    int gi = idxq[((size_t)b * N_ + n) * 16 + kq];
    _Float16* xr = X0 + row * 160;
    if (part == 0) {
      const float* pp = wxyz + ((size_t)b * N_ + n) * 3;
      const float* gp = fxyz + ((size_t)b * N_ + gi) * 3;
      float px = pp[0], py = pp[1], pz = pp[2];
      float gx = gp[0], gy = gp[1], gz = gp[2];
      float dx = gx - px, dy = gy - py, dz = gz - pz;
      float e = sqrtf(dx * dx + dy * dy + dz * dz + 1e-20f);
      xr[0] = (_Float16)px; xr[1] = (_Float16)py; xr[2] = (_Float16)pz;
      xr[3] = (_Float16)gx; xr[4] = (_Float16)gy; xr[5] = (_Float16)gz;
      xr[6] = (_Float16)dx; xr[7] = (_Float16)dy; xr[8] = (_Float16)dz;
      xr[9] = (_Float16)e;
      const float* wrow = wpts + ((size_t)b * N_ + n) * C_;
      for (int c = 0; c < C_; ++c) xr[10 + c] = (_Float16)wrow[c];
      for (int c = 138; c < 160; ++c) xr[c] = (_Float16)0.f;
    } else {
      const float* frow = fpts + ((size_t)b * N_ + gi) * C_;
      for (int c = 0; c < C_; ++c) xr[74 + c] = (_Float16)frow[c];
    }
  }
  wait_async_lds();
  __syncthreads();
  // enc1: X0 cols 0..31 (pad channels hit zero weights) -> bufC cols 0..63
  wave_gemm<1, 4, true>(X0, 160, bufW + kWOFF[3], st + 3 * 256, st + 3 * 256 + 128,
                        bufC, nullptr, 128, 0, lane, wave);
  __syncthreads();
  // mlp1_0: 160(K) -> 128
  wave_gemm<5, 8, true>(X0, 160, bufW + kWOFF[0], st + 0 * 256, st + 0 * 256 + 128,
                        bufA, nullptr, 128, 0, lane, wave);
  __syncthreads();
  // mlp1_1: 128 -> 64 (into X0 region as h2)
  wave_gemm<4, 4, true>(bufA, 128, bufW + kWOFF[1], st + 1 * 256, st + 1 * 256 + 128,
                        h2, nullptr, 64, 0, lane, wave);
  __syncthreads();
  // mlp1_2: 64 -> 64 (h3, into bufC cols 64..127)
  wave_gemm<2, 4, true>(h2, 64, bufW + kWOFF[2], st + 2 * 256, st + 2 * 256 + 128,
                        bufC, nullptr, 128, 64, lane, wave);
  __syncthreads();
  // mlp2_0: 128 -> 128
  wave_gemm<4, 8, true>(bufC, 128, bufW + kWOFF[4], st + 4 * 256, st + 4 * 256 + 128,
                        bufA, nullptr, 128, 0, lane, wave);
  __syncthreads();
  // mlp2_1: 128 -> 64 logits (f32)
  wave_gemm<4, 4, false>(bufA, 128, bufW + kWOFF[5], st + 5 * 256, st + 5 * 256 + 128,
                         nullptr, bufS, 64, 0, lane, wave);
  __syncthreads();
  {  // softmax over the 16 neighbors + weighted sum of h3 -> feat1[b,n,64]
    int qq = wave;
    int n = n0 + qq;
    for (int cc = 0; cc < 2; ++cc) {
      int ch = (cc << 5) + lane;
      float mx = -3.0e38f;
#pragma unroll
      for (int k = 0; k < 16; ++k) mx = fmaxf(mx, bufS[(qq * 16 + k) * 64 + ch]);
      float se = 0.f, ws = 0.f;
#pragma unroll
      for (int k = 0; k < 16; ++k) {
        float e = __expf(bufS[(qq * 16 + k) * 64 + ch] - mx);
        se += e;
        ws += e * (float)bufC[(qq * 16 + k) * 128 + 64 + ch];
      }
      feat1[((size_t)b * N_ + n) * C_ + ch] = ws / se;
    }
  }
}

// ---------------------------------------------------------------------------
// Stage 2: self grouping + enc2 + mlp3 chain (mlp2 BN stats) + softmax.
// ---------------------------------------------------------------------------
__global__ void __launch_bounds__(256)
stage2_kernel(const float* __restrict__ wxyz, const float* __restrict__ wpts,
              const float* __restrict__ feat1, const int* __restrict__ idxs,
              const _Float16* __restrict__ Wp, const float* __restrict__ st,
              float* __restrict__ out) {
  extern __shared__ char smem[];
  _Float16* bufXYZ = (_Float16*)smem;                          // 128x32 f16 (8192B)
  _Float16* bufC2  = (_Float16*)(smem + 8192);                 // 128x192 f16 (49152B)
  _Float16* bufA2  = (_Float16*)(smem + 8192 + 49152);         // 128x128 f16 (32768B)
  float*    bufS   = (float*)(smem + 8192 + 49152 + 32768);    // 128x64 f32
  _Float16* bufW   = (_Float16*)(smem + kS2_W_LDSOFF);         // staged weights L6..L8

  const int tid = threadIdx.x;
  const int lane = tid & 31, wave = tid >> 5;
  const int b = blockIdx.x >> 9;
  const int n0 = (blockIdx.x & 511) << 3;

  // stage weights for layers 6..8 (contiguous block starting at kWOFF[6])
  async_stage_weights(kS2_W_LDSOFF, Wp + kWOFF[6], (int)kS2_W_BYTES, tid);

  {  // build xyz_cat2 and c2 = [enc2 | pts_new | pts_g]
    int row = tid & 127, part = tid >> 7;
    int qq = row >> 4, kk = row & 15;
    int n = n0 + qq;
    int gi = idxs[((size_t)b * N_ + n) * 16 + kk];
    if (part == 0) {
      const float* pp = wxyz + ((size_t)b * N_ + n) * 3;
      const float* gp = wxyz + ((size_t)b * N_ + gi) * 3;
      float px = pp[0], py = pp[1], pz = pp[2];
      float gx = gp[0], gy = gp[1], gz = gp[2];
      float dx = gx - px, dy = gy - py, dz = gz - pz;
      float e = sqrtf(dx * dx + dy * dy + dz * dz + 1e-20f);
      _Float16* xr = bufXYZ + row * 32;
      xr[0] = (_Float16)px; xr[1] = (_Float16)py; xr[2] = (_Float16)pz;
      xr[3] = (_Float16)gx; xr[4] = (_Float16)gy; xr[5] = (_Float16)gz;
      xr[6] = (_Float16)dx; xr[7] = (_Float16)dy; xr[8] = (_Float16)dz;
      xr[9] = (_Float16)e;
      for (int c = 10; c < 32; ++c) xr[c] = (_Float16)0.f;
      const float* wrow = wpts + ((size_t)b * N_ + n) * C_;
      for (int c = 0; c < C_; ++c) bufC2[row * 192 + 64 + c] = (_Float16)wrow[c];
    } else {
      const float* frow = feat1 + ((size_t)b * N_ + gi) * C_;
      for (int c = 0; c < C_; ++c) bufC2[row * 192 + 128 + c] = (_Float16)frow[c];
    }
  }
  wait_async_lds();
  __syncthreads();
  // enc2: 32(K) -> 64 (into bufC2 cols 0..63)
  wave_gemm<1, 4, true>(bufXYZ, 32, bufW, st + 6 * 256, st + 6 * 256 + 128,
                        bufC2, nullptr, 192, 0, lane, wave);
  __syncthreads();
  // mlp3_0: 192 -> 128 (BN from mlp2[0])
  wave_gemm<6, 8, true>(bufC2, 192, bufW + (kWOFF[7] - kWOFF[6]),
                        st + 7 * 256, st + 7 * 256 + 128,
                        bufA2, nullptr, 128, 0, lane, wave);
  __syncthreads();
  // mlp3_1: 128 -> 64 logits (BN from mlp2[1])
  wave_gemm<4, 4, false>(bufA2, 128, bufW + (kWOFF[8] - kWOFF[6]),
                         st + 8 * 256, st + 8 * 256 + 128,
                         nullptr, bufS, 64, 0, lane, wave);
  __syncthreads();
  {  // softmax over 16 + weighted sum of gathered feat1 -> out[b,n,64]
    int qq = wave;
    int n = n0 + qq;
    for (int cc = 0; cc < 2; ++cc) {
      int ch = (cc << 5) + lane;
      float mx = -3.0e38f;
#pragma unroll
      for (int k = 0; k < 16; ++k) mx = fmaxf(mx, bufS[(qq * 16 + k) * 64 + ch]);
      float se = 0.f, ws = 0.f;
#pragma unroll
      for (int k = 0; k < 16; ++k) {
        float e = __expf(bufS[(qq * 16 + k) * 64 + ch] - mx);
        se += e;
        ws += e * (float)bufC2[(qq * 16 + k) * 192 + 128 + ch];
      }
      out[((size_t)b * N_ + n) * C_ + ch] = ws / se;
    }
  }
}

// ---------------------------------------------------------------------------
// Host launcher. Input order follows setup_inputs() insertion order with the
// params pytree flattened in insertion order:
//   0 warped_xyz, 1 f2_xyz, 2 warped_points, 3 f2_points,
//   mlp1[0..2]: {W,b,g,beta,rm,rv} -> 4..21, encode1 -> 22..27,
//   mlp2[0..1] -> 28..39, encode2 -> 40..45, mlp3[0..1]: {W,b} -> 46..49
// ---------------------------------------------------------------------------
extern "C" void kernel_launch(void* const* d_in, const int* in_sizes, int n_in,
                              void* d_out, int out_size, void* d_ws, size_t ws_size,
                              hipStream_t stream) {
  (void)in_sizes; (void)n_in; (void)out_size; (void)ws_size;
  const float* wxyz = (const float*)d_in[0];
  const float* fxyz = (const float*)d_in[1];
  const float* wpts = (const float*)d_in[2];
  const float* fpts = (const float*)d_in[3];

  char* ws = (char*)d_ws;
  _Float16* Wpack = (_Float16*)(ws + WS_WPACK);
  float*    st    = (float*)(ws + WS_ST);
  int*      idxq  = (int*)(ws + WS_IDXQ);
  int*      idxs  = (int*)(ws + WS_IDXS);
  float*    feat1 = (float*)(ws + WS_FEAT1);

  struct L { int w, b, g, be, rm, rv, cin, cout, kpad; };
  const L T[9] = {
      {4, 5, 6, 7, 8, 9, 138, 128, 160},        // mlp1_0
      {10, 11, 12, 13, 14, 15, 128, 64, 128},   // mlp1_1
      {16, 17, 18, 19, 20, 21, 64, 64, 64},     // mlp1_2
      {22, 23, 24, 25, 26, 27, 10, 64, 32},     // encode1
      {28, 29, 30, 31, 32, 33, 128, 128, 128},  // mlp2_0
      {34, 35, 36, 37, 38, 39, 128, 64, 128},   // mlp2_1
      {40, 41, 42, 43, 44, 45, 10, 64, 32},     // encode2
      {46, 47, 30, 31, 32, 33, 192, 128, 192},  // mlp3_0 (mlp2_0 BN stats)
      {48, 49, 36, 37, 38, 39, 128, 64, 128},   // mlp3_1 (mlp2_1 BN stats)
  };
  for (int i = 0; i < 9; ++i) {
    const L& l = T[i];
    int total = l.kpad * l.cout;
    prep_layer<<<(total + 255) / 256, 256, 0, stream>>>(
        (const float*)d_in[l.w], (const float*)d_in[l.b], (const float*)d_in[l.g],
        (const float*)d_in[l.be], (const float*)d_in[l.rm], (const float*)d_in[l.rv],
        l.cin, l.cout, l.kpad, Wpack + kWOFF[i], st + i * 256, st + i * 256 + 128);
  }

  knn_kernel<<<B_ * (N_ / 256), 256, 0, stream>>>(wxyz, fxyz, idxq);
  knn_kernel<<<B_ * (N_ / 256), 256, 0, stream>>>(wxyz, wxyz, idxs);

  hipFuncSetAttribute((const void*)stage1_kernel,
                      hipFuncAttributeMaxDynamicSharedMemorySize, (int)kSMEM1);
  hipFuncSetAttribute((const void*)stage2_kernel,
                      hipFuncAttributeMaxDynamicSharedMemorySize, (int)kSMEM2);

  stage1_kernel<<<B_ * (N_ / 8), 256, kSMEM1, stream>>>(wxyz, fxyz, wpts, fpts, idxq,
                                                        Wpack, st, feat1);
  stage2_kernel<<<B_ * (N_ / 8), 256, kSMEM2, stream>>>(wxyz, wpts, feat1, idxs,
                                                        Wpack, st, (float*)d_out);
}
